// RNNModel_41291815584251
// MI455X (gfx1250) — matrix operand order repaired and checked by reading, not compiled, not run on hardware
//
#include <hip/hip_runtime.h>

// ---------------------------------------------------------------------------
// CDNA5 (gfx1250) LSTM language model forward.
// - All GEMMs via v_wmma_f32_16x16x32_f16, operands pre-packed in the ISA
//   16-bit WMMA lane layouts (f16).
// - LSTM cell fully fused into the gates GEMM (gate-strided n-tiles).
// - Decoder stages B through LDS with global_load_async_to_lds_b128.
// ---------------------------------------------------------------------------

typedef __attribute__((ext_vector_type(16))) _Float16 v16h;
typedef __attribute__((ext_vector_type(8)))  float    v8f;

#define SEQL 64
#define BSZ  32
#define NTOK 50000
#define NINP 512
#define NHID 1024
#define NGATE (4 * NHID)                     // 4096
#define FLAT (SEQL * BSZ)                    // 2048
#define HSZ  (BSZ * NHID)                    // 32768
#define DEC_ELEMS ((long long)FLAT * NTOK)   // 102,400,000

// Fast device transcendentals (v_exp_f32 based; exact at saturation).
static __device__ __forceinline__ float fast_sigmoid(float x) {
    return 1.0f / (1.0f + __expf(-x));
}
static __device__ __forceinline__ float fast_tanh(float x) {
    return 1.0f - 2.0f / (__expf(2.0f * x) + 1.0f);
}

static __device__ __forceinline__ v8f wmma16x16x32(v16h a, v16h b, v8f c) {
    return __builtin_amdgcn_wmma_f32_16x16x32_f16(
        false, a, false, b, (short)0, c, false, false);
}

// Packed tiles (A or B): tile = 512 f16 = 32 lanes x 16 halfs, lane-contiguous.
static __device__ __forceinline__ v16h load_tile(const _Float16* __restrict__ P,
                                                 long long tile, int lane) {
    return *(const v16h*)(P + tile * 512 + lane * 16);
}

// Index of element (m,k) of a row-major [M,K] matrix inside the packed-A
// tile array (ISA 16-bit A 16x32 layout). KT = K/32.
static __device__ __forceinline__ long long a_pack_index(int m, int k, int KT) {
    int mt = m >> 4, kt = k >> 5, ko = k & 31;
    int lane = (m & 15) + (((ko >> 3) & 1) << 4);
    int e = ((ko >> 4) << 3) + (ko & 7);
    return ((long long)(mt * KT + kt) * 32 + lane) * 16 + e;
}

// ---------------------------------------------------------------------------
// Pack W[N,K] (f32; GEMM uses B = W^T) into the ISA 16-bit B 32x16 layout.
// ---------------------------------------------------------------------------
__global__ void pack_b_kernel(const float* __restrict__ W, _Float16* __restrict__ dst,
                              int N, int K) {
    long long idx = (long long)blockIdx.x * blockDim.x + threadIdx.x;  // half-pairs
    int KT = K >> 5;
    long long total = (long long)(N >> 4) * KT * 256;
    if (idx >= total) return;
    int j    = (int)(idx & 7);
    int lane = (int)((idx >> 3) & 31);
    long long t = idx >> 8;
    int kt = (int)(t % KT);
    int nt = (int)(t / KT);
    int n = nt * 16 + (lane & 15);
    int k = kt * 32 + ((lane >> 4) << 3) + (j < 4 ? 2 * j : 8 + 2 * j);
    const float* src = W + (long long)n * K + k;
    dst[idx * 2]     = (_Float16)src[0];
    dst[idx * 2 + 1] = (_Float16)src[1];
}

// Embedding gather fused with packed-A conversion: dst is packed [2048,512].
__global__ void embed_pack_kernel(const int* __restrict__ ids,
                                  const float* __restrict__ emb_w,
                                  _Float16* __restrict__ dst) {
    long long idx = (long long)blockIdx.x * blockDim.x + threadIdx.x;
    if (idx >= (long long)FLAT * NINP) return;
    int m = (int)(idx >> 9);
    int k = (int)(idx & (NINP - 1));
    float v = emb_w[(long long)ids[m] * NINP + k];
    dst[a_pack_index(m, k, NINP / 32)] = (_Float16)v;
}

// Initial packed-h conversion: hx[layer][32][1024] f32 -> packed-A f16.
__global__ void hpack_init_kernel(const float* __restrict__ hx,
                                  _Float16* __restrict__ hP0,
                                  _Float16* __restrict__ hP1) {
    int idx = blockIdx.x * blockDim.x + threadIdx.x;
    if (idx >= 2 * HSZ) return;
    int layer = idx / HSZ;
    int r = idx % HSZ;
    int b = r >> 10;
    int n = r & (NHID - 1);
    _Float16* dst = layer ? hP1 : hP0;
    dst[a_pack_index(b, n, NHID / 32)] = (_Float16)hx[idx];
}

__global__ void bias_combine_kernel(const float* __restrict__ a, const float* __restrict__ b,
                                    float* __restrict__ o, int n) {
    int i = blockIdx.x * blockDim.x + threadIdx.x;
    if (i < n) o[i] = a[i] + b[i];
}

__global__ void copyf_kernel(const float* __restrict__ s, float* __restrict__ d, long long n) {
    long long i = (long long)blockIdx.x * blockDim.x + threadIdx.x;
    if (i < n) d[i] = s[i];
}

// ---------------------------------------------------------------------------
// Fully fused LSTM cell. One wave handles hidden columns n0..n0+15 for all
// 32 batch rows: its 4 n-tiles are GATE-STRIDED (nt = q, q+64, q+128, q+192),
// so acc[mi][0..3] hold the i/f/g/o gates for the SAME (batch, hidden)
// elements. The cell nonlinearity fuses in-register; no gates buffer.
// h is ping-ponged: reads APh (old packed h), writes hOutP (new packed h).
// ---------------------------------------------------------------------------
template <int KTX>
__global__ void __launch_bounds__(256)
lstm_cell_kernel(const _Float16* __restrict__ APx, int mtx0,
                 const _Float16* __restrict__ APh,
                 const _Float16* __restrict__ PWih, const _Float16* __restrict__ PWhh,
                 const float* __restrict__ bias,
                 float* __restrict__ cstate, float* __restrict__ hstate,
                 _Float16* __restrict__ hOutP,
                 _Float16* __restrict__ seqPacked, int seqRow0) {
    const int lane = threadIdx.x & 31;
    const int q = blockIdx.x * (blockDim.x >> 5) + (threadIdx.x >> 5);  // 0..63
    const int col = lane & 15;
    const int n = q * 16 + col;          // hidden column owned by this lane

    v8f acc[2][4];
#pragma unroll
    for (int t = 0; t < 4; ++t) {
        float bv = bias[t * NHID + n];
#pragma unroll
        for (int i = 0; i < 8; ++i) { acc[0][t][i] = bv; acc[1][t][i] = bv; }
    }

    for (int kt = 0; kt < KTX; ++kt) {
        v16h a0 = load_tile(APx, (long long)(mtx0 + 0) * KTX + kt, lane);
        v16h a1 = load_tile(APx, (long long)(mtx0 + 1) * KTX + kt, lane);
#pragma unroll
        for (int t = 0; t < 4; ++t) {
            v16h b = load_tile(PWih, (long long)(q + t * 64) * KTX + kt, lane);
            acc[0][t] = wmma16x16x32(a0, b, acc[0][t]);
            acc[1][t] = wmma16x16x32(a1, b, acc[1][t]);
        }
    }
    const int KTH = NHID / 32;
    for (int kt = 0; kt < KTH; ++kt) {
        v16h a0 = load_tile(APh, kt, lane);
        v16h a1 = load_tile(APh, KTH + kt, lane);
#pragma unroll
        for (int t = 0; t < 4; ++t) {
            v16h b = load_tile(PWhh, (long long)(q + t * 64) * KTH + kt, lane);
            acc[0][t] = wmma16x16x32(a0, b, acc[0][t]);
            acc[1][t] = wmma16x16x32(a1, b, acc[1][t]);
        }
    }

    // Fused LSTM cell: element i of each acc tile is (batch=mi*16+rb+i, n).
    const int rb = (lane >> 4) << 3;
#pragma unroll
    for (int mi = 0; mi < 2; ++mi) {
#pragma unroll
        for (int i = 0; i < 8; ++i) {
            const int b = mi * 16 + rb + i;
            const int idx = b * NHID + n;
            float ig = fast_sigmoid(acc[mi][0][i]);
            float fg = fast_sigmoid(acc[mi][1][i]);
            float gg = fast_tanh(acc[mi][2][i]);
            float og = fast_sigmoid(acc[mi][3][i]);
            float c2 = fg * cstate[idx] + ig * gg;
            float h2 = og * fast_tanh(c2);
            cstate[idx] = c2;
            hstate[idx] = h2;
            _Float16 h16 = (_Float16)h2;
            hOutP[a_pack_index(b, n, NHID / 32)] = h16;
            if (seqPacked)
                seqPacked[a_pack_index(seqRow0 + b, n, NHID / 32)] = h16;
        }
    }
}

// ---------------------------------------------------------------------------
// Predictor: P[2048,512] = tanh((seq@Wp^T + bp) / sqrt(1+eps)),
// stored directly in packed-A layout (f16) for the decoder.
// ---------------------------------------------------------------------------
__global__ void __launch_bounds__(256)
predictor_kernel(const _Float16* __restrict__ AP, const _Float16* __restrict__ PB,
                 const float* __restrict__ bp, _Float16* __restrict__ Ppacked) {
    const float invs = 0.99999500003749963f;  // 1/sqrt(1 + 1e-5)
    const int lane = threadIdx.x & 31;
    const int wave = blockIdx.x * (blockDim.x >> 5) + (threadIdx.x >> 5);
    const int mt  = wave >> 3;        // 128 m-tiles
    const int nt0 = (wave & 7) * 4;   // 32 n-tiles, 4 per wave
    const int col = lane & 15;

    v8f acc[4];
#pragma unroll
    for (int t = 0; t < 4; ++t) {
        float bv = bp[nt0 * 16 + t * 16 + col];
#pragma unroll
        for (int i = 0; i < 8; ++i) acc[t][i] = bv;
    }
    const int KT = NHID / 32;   // 32
    for (int kt = 0; kt < KT; ++kt) {
        v16h a = load_tile(AP, (long long)mt * KT + kt, lane);
#pragma unroll
        for (int t = 0; t < 4; ++t) {
            v16h b = load_tile(PB, (long long)(nt0 + t) * KT + kt, lane);
            acc[t] = wmma16x16x32(a, b, acc[t]);
        }
    }
    const int rbase = mt * 16 + ((lane >> 4) << 3);
#pragma unroll
    for (int t = 0; t < 4; ++t) {
        const int c0 = nt0 * 16 + t * 16 + col;
#pragma unroll
        for (int i = 0; i < 8; ++i) {
            float v = fast_tanh(acc[t][i] * invs);
            Ppacked[a_pack_index(rbase + i, c0, NINP / 32)] = (_Float16)v;
        }
    }
}

// ---------------------------------------------------------------------------
// Decoder: Out[2048,50000] = P@Wd^T + bd.
// Block = 8 waves; stages its 5 n-tiles x 16 k-tiles of packed B (80 KB,
// contiguous) into LDS via async global->LDS copies (ASYNCcnt), then each
// wave computes 2 m-tiles x 5 n-tiles (160 wmma/wave).
// ---------------------------------------------------------------------------
#define DEC_KT (NINP / 32)                 // 16
#define DEC_LDS_BYTES (5 * DEC_KT * 1024)  // 81920 B
__global__ void __launch_bounds__(256)
decoder_kernel(const _Float16* __restrict__ AP, const _Float16* __restrict__ PB,
               const float* __restrict__ bd, float* __restrict__ Out) {
    extern __shared__ uint4 ldsv[];
    _Float16* lds = (_Float16*)ldsv;

    const int tid  = threadIdx.x;
    const int lane = tid & 31;
    const int w    = tid >> 5;             // wave in block: 0..7
    const int mg   = blockIdx.x / 625;     // 8 m-groups of 256 rows
    const int ng   = blockIdx.x % 625;     // 625 n-groups of 80 cols
    const int nt0  = ng * 5;
    const int col  = lane & 15;

    // Async stage of B into LDS (tiles nt0..nt0+4 are contiguous in PB).
    {
        const char* gsrc = (const char*)(PB + (long long)nt0 * DEC_KT * 512);
        unsigned lbase = (unsigned)(size_t)(void*)lds;   // LDS byte offset
#pragma unroll 4
        for (int off = tid * 16; off < DEC_LDS_BYTES; off += 256 * 16) {
            unsigned l = lbase + (unsigned)off;
            const char* g = gsrc + off;
            asm volatile("global_load_async_to_lds_b128 %0, %1, off"
                         :: "v"(l), "v"(g) : "memory");
        }
        asm volatile("s_wait_asynccnt 0x0" ::: "memory");
    }
    __syncthreads();

    const int mtb = mg * 16 + w * 2;       // two m-tiles per wave
    v8f acc[2][5];
#pragma unroll
    for (int t = 0; t < 5; ++t) {
        float bv = bd[nt0 * 16 + t * 16 + col];
#pragma unroll
        for (int i = 0; i < 8; ++i) { acc[0][t][i] = bv; acc[1][t][i] = bv; }
    }

    for (int kt = 0; kt < DEC_KT; ++kt) {
        v16h a0 = load_tile(AP, (long long)(mtb + 0) * DEC_KT + kt, lane);
        v16h a1 = load_tile(AP, (long long)(mtb + 1) * DEC_KT + kt, lane);
#pragma unroll
        for (int t = 0; t < 5; ++t) {
            v16h b = *(const v16h*)(lds + ((long long)(t * DEC_KT + kt) * 32 + lane) * 16);
            acc[0][t] = wmma16x16x32(a0, b, acc[0][t]);
            acc[1][t] = wmma16x16x32(a1, b, acc[1][t]);
        }
    }

    const int rb = (lane >> 4) << 3;
#pragma unroll
    for (int mi = 0; mi < 2; ++mi) {
        const int row0 = (mtb + mi) * 16 + rb;
#pragma unroll
        for (int t = 0; t < 5; ++t) {
            const int c0 = nt0 * 16 + t * 16 + col;
#pragma unroll
            for (int i = 0; i < 8; ++i)
                Out[(long long)(row0 + i) * NTOK + c0] = acc[mi][t][i];
        }
    }
}

// ---------------------------------------------------------------------------
extern "C" void kernel_launch(void* const* d_in, const int* in_sizes, int n_in,
                              void* d_out, int out_size, void* d_ws, size_t ws_size,
                              hipStream_t stream) {
    const int*   ids   = (const int*)d_in[0];
    const float* hx    = (const float*)d_in[1];
    const float* cx    = (const float*)d_in[2];
    const float* emb_w = (const float*)d_in[3];
    const float* Wih0  = (const float*)d_in[4];
    const float* Whh0  = (const float*)d_in[5];
    const float* bih0  = (const float*)d_in[6];
    const float* bhh0  = (const float*)d_in[7];
    const float* Wih1  = (const float*)d_in[8];
    const float* Whh1  = (const float*)d_in[9];
    const float* bih1  = (const float*)d_in[10];
    const float* bhh1  = (const float*)d_in[11];
    const float* Wp    = (const float*)d_in[12];
    const float* bp    = (const float*)d_in[13];
    const float* Wd    = (const float*)d_in[14];
    const float* bd    = (const float*)d_in[15];
    float* out = (float*)d_out;

    // ---- workspace layout ----
    size_t off = 0;
    auto alloc = [&](size_t bytes) {
        off = (off + 255) & ~(size_t)255;
        size_t o = off;
        off += bytes;
        return o;
    };
    char* ws = (char*)d_ws;
    size_t o_pWih0 = alloc((size_t)NGATE * NINP * 2);
    size_t o_pWhh0 = alloc((size_t)NGATE * NHID * 2);
    size_t o_pWih1 = alloc((size_t)NGATE * NHID * 2);
    size_t o_pWhh1 = alloc((size_t)NGATE * NHID * 2);
    size_t o_pWp   = alloc((size_t)NINP * NHID * 2);
    size_t o_pWd   = alloc((size_t)NTOK * NINP * 2);
    size_t o_embP  = alloc((size_t)FLAT * NINP * 2);
    size_t o_seqP  = alloc((size_t)FLAT * NHID * 2);
    size_t o_hbuf  = alloc((size_t)2 * HSZ * 4);     // f32 h states
    size_t o_cbuf  = alloc((size_t)2 * HSZ * 4);     // f32 c states
    size_t o_hP    = alloc((size_t)4 * HSZ * 2);     // packed h ping-pong x2 layers
    size_t o_b0    = alloc((size_t)NGATE * 4);
    size_t o_b1    = alloc((size_t)NGATE * 4);
    size_t o_pP    = alloc((size_t)FLAT * NINP * 2);
    if (off > ws_size) return;

    _Float16* pWih0 = (_Float16*)(ws + o_pWih0);
    _Float16* pWhh0 = (_Float16*)(ws + o_pWhh0);
    _Float16* pWih1 = (_Float16*)(ws + o_pWih1);
    _Float16* pWhh1 = (_Float16*)(ws + o_pWhh1);
    _Float16* pWp   = (_Float16*)(ws + o_pWp);
    _Float16* pWd   = (_Float16*)(ws + o_pWd);
    _Float16* embP  = (_Float16*)(ws + o_embP);
    _Float16* seqP  = (_Float16*)(ws + o_seqP);
    float* hbuf  = (float*)(ws + o_hbuf);
    float* cbuf  = (float*)(ws + o_cbuf);
    _Float16* hP = (_Float16*)(ws + o_hP);  // [L0 ping, L0 pong, L1 ping, L1 pong]
    float* bias0 = (float*)(ws + o_b0);
    float* bias1 = (float*)(ws + o_b1);
    _Float16* pP = (_Float16*)(ws + o_pP);

    auto grid1d = [](long long n, int bs) { return (unsigned)((n + bs - 1) / bs); };

    // ---- pack weights to f16 WMMA-B layout ----
    {
        long long t;
        t = (long long)(NGATE / 16) * (NINP / 32) * 256;
        pack_b_kernel<<<grid1d(t, 256), 256, 0, stream>>>(Wih0, pWih0, NGATE, NINP);
        t = (long long)(NGATE / 16) * (NHID / 32) * 256;
        pack_b_kernel<<<grid1d(t, 256), 256, 0, stream>>>(Whh0, pWhh0, NGATE, NHID);
        pack_b_kernel<<<grid1d(t, 256), 256, 0, stream>>>(Wih1, pWih1, NGATE, NHID);
        pack_b_kernel<<<grid1d(t, 256), 256, 0, stream>>>(Whh1, pWhh1, NGATE, NHID);
        t = (long long)(NINP / 16) * (NHID / 32) * 256;
        pack_b_kernel<<<grid1d(t, 256), 256, 0, stream>>>(Wp, pWp, NINP, NHID);
        t = (long long)(NTOK / 16) * (NINP / 32) * 256;
        pack_b_kernel<<<grid1d(t, 256), 256, 0, stream>>>(Wd, pWd, NTOK, NINP);
    }

    // ---- biases, embedding (gather + pack), state init ----
    bias_combine_kernel<<<grid1d(NGATE, 256), 256, 0, stream>>>(bih0, bhh0, bias0, NGATE);
    bias_combine_kernel<<<grid1d(NGATE, 256), 256, 0, stream>>>(bih1, bhh1, bias1, NGATE);
    embed_pack_kernel<<<grid1d((long long)FLAT * NINP, 256), 256, 0, stream>>>(ids, emb_w, embP);
    copyf_kernel<<<grid1d(2 * HSZ, 256), 256, 0, stream>>>(hx, hbuf, 2 * HSZ);
    copyf_kernel<<<grid1d(2 * HSZ, 256), 256, 0, stream>>>(cx, cbuf, 2 * HSZ);
    hpack_init_kernel<<<grid1d(2 * HSZ, 256), 256, 0, stream>>>(hx, hP, hP + 2 * HSZ);

    float* h0 = hbuf;
    float* h1 = hbuf + HSZ;
    float* c0 = cbuf;
    float* c1 = cbuf + HSZ;

    // ---- sequential LSTM: one fused kernel per layer-step ----
    for (int s = 0; s < SEQL; ++s) {
        const int pr = s & 1;             // ping-pong parity
        _Float16* h0r = hP + (pr ? HSZ : 0);
        _Float16* h0w = hP + (pr ? 0 : HSZ);
        _Float16* h1r = hP + 2 * HSZ + (pr ? HSZ : 0);
        _Float16* h1w = hP + 2 * HSZ + (pr ? 0 : HSZ);
        lstm_cell_kernel<NINP / 32><<<8, 256, 0, stream>>>(
            embP, 2 * s, h0r, pWih0, pWhh0, bias0, c0, h0, h0w,
            (_Float16*)nullptr, 0);
        lstm_cell_kernel<NHID / 32><<<8, 256, 0, stream>>>(
            h0w, 0, h1r, pWih1, pWhh1, bias1, c1, h1, h1w,
            seqP, s * BSZ);
    }

    // ---- predictor (Linear + BN(eval) + Tanh), output packed for decoder ----
    predictor_kernel<<<128, 256, 0, stream>>>(seqP, pWp, bp, pP);

    // ---- decoder GEMM with async-LDS-staged B, straight into d_out ----
    decoder_kernel<<<5000, 256, DEC_LDS_BYTES, stream>>>(pP, pWd, bd, out);

    // ---- final hidden / cell states ----
    copyf_kernel<<<grid1d(2 * HSZ, 256), 256, 0, stream>>>(
        hbuf, out + DEC_ELEMS, 2 * HSZ);
    copyf_kernel<<<grid1d(2 * HSZ, 256), 256, 0, stream>>>(
        cbuf, out + DEC_ELEMS + 2 * HSZ, 2 * HSZ);
}
